// MPNN_enn_edge_51505247813804
// MI455X (gfx1250) — matrix operand order repaired and checked by reading, not compiled, not run on hardware
//
#include <hip/hip_runtime.h>

// ---------------- types for WMMA / NT loads ----------------
typedef __attribute__((ext_vector_type(16))) __bf16 v16bf;
typedef __attribute__((ext_vector_type(8)))  float  v8f;
typedef __attribute__((ext_vector_type(4)))  unsigned int u32x4;

union FragU { uint4 q[2]; v16bf v; };

// ---------------- bf16 helpers ----------------
__device__ __forceinline__ unsigned short f2b(float f) {
    unsigned int u = __float_as_uint(f);
    u += 0x7FFFu + ((u >> 16) & 1u);   // round-to-nearest-even
    return (unsigned short)(u >> 16);
}
__device__ __forceinline__ float b2f_lo(unsigned int u) { return __uint_as_float(u << 16); }
__device__ __forceinline__ float b2f_hi(unsigned int u) { return __uint_as_float(u & 0xFFFF0000u); }

// ---------------- problem constants (from reference) ----------------
#define NN   2000
#define EE   64000
#define EDD  16
#define ENHH 128
#define HH   64
#define HC   4096    // H*H
#define TSTEPS 8

// ================================================================
// Swizzle W2 (128x4096 f32) into bf16 WMMA-B fragment order.
//   w2s[((nt*4 + ks)*32 + lane)*16 + p]
//   k = ks*32 + ((lane&16)?8:0) + (p<8 ? p : p+8) ,  n = nt*16 + (lane&15)
// ================================================================
__global__ void kw2_swizzle(const float* __restrict__ W2, unsigned short* __restrict__ w2s) {
    int tid = blockIdx.x * 256 + threadIdx.x;            // < 524288
    if (tid >= 256 * 4 * 32 * 16) return;
    int p    = tid & 15;
    int lane = (tid >> 4) & 31;
    int ks   = (tid >> 9) & 3;
    int nt   = tid >> 11;
    int klo  = (lane & 16) ? 8 : 0;
    int k    = ks * 32 + klo + (p < 8 ? p : p + 8);
    int n    = nt * 16 + (lane & 15);
    w2s[tid] = f2b(W2[(size_t)k * HC + n]);
}

// ================================================================
// K1: h1 = relu(ef @ W1 + b1) -> bf16 (E x 128). One block per edge.
// ================================================================
__global__ void k1_h1(const float* __restrict__ edge_data,
                      const int*   __restrict__ edges,
                      const float* __restrict__ W1,
                      const float* __restrict__ b1,
                      unsigned short* __restrict__ h1b) {
    __shared__ float ef[EDD];
    int e = blockIdx.x;
    int t = threadIdx.x;                    // 0..127
    int s  = edges[2 * e + 0];
    int tg = edges[2 * e + 1];
    if (t < EDD) ef[t] = edge_data[((size_t)s * NN + tg) * EDD + t];
    __syncthreads();
    float a = b1[t];
#pragma unroll
    for (int d = 0; d < EDD; ++d) a += ef[d] * W1[d * ENHH + t];
    a = a > 0.f ? a : 0.f;
    h1b[(size_t)e * ENHH + t] = f2b(a);
}

// ================================================================
// K2: A = relu(h1 @ W2 + b2) -> bf16 (E x 4096) via WMMA.
// 128 threads = 4 waves per block; block owns 16 edges.
// NT stores: A is a stream, keep it out of L2 retention.
// ================================================================
__global__ void k2_edge_matrices(const unsigned short* __restrict__ h1b,
                                 const unsigned short* __restrict__ w2s,
                                 const float* __restrict__ b2,
                                 unsigned short* __restrict__ Ab) {
    const int lane = threadIdx.x & 31;
    const int wave = threadIdx.x >> 5;
    const int e0   = blockIdx.x * 16;
    if (e0 >= EE) return;

    const int m  = lane & 15;
    const int kb = (lane < 16) ? 0 : 8;       // per-half K base (ISA A-layout)

    FragU afr[4];
    const size_t rowbase = (size_t)(e0 + m) * ENHH;
#pragma unroll
    for (int ks = 0; ks < 4; ++ks) {
        const uint4* p = (const uint4*)(h1b + rowbase + ks * 32 + kb);
        afr[ks].q[0] = p[0];   // K = ks*32 + kb .. +7
        afr[ks].q[1] = p[2];   // K = ks*32 + 16+kb .. +7
    }

    const int n  = lane & 15;
    const int mo = (lane < 16) ? 0 : 8;

    for (int nti = 0; nti < 64; ++nti) {
        const int nt = wave * 64 + nti;
        const int c0 = nt * 16;
        v8f c = {0.f, 0.f, 0.f, 0.f, 0.f, 0.f, 0.f, 0.f};
#pragma unroll
        for (int ks = 0; ks < 4; ++ks) {
            FragU bfr;
            const uint4* bp = (const uint4*)(w2s + ((size_t)(nt * 4 + ks) * 32 + lane) * 16);
            bfr.q[0] = bp[0];
            bfr.q[1] = bp[1];
            c = __builtin_amdgcn_wmma_f32_16x16x32_bf16(
                    false, afr[ks].v, false, bfr.v, (short)0, c, false, false);
        }
        const float bias = b2[c0 + n];
#pragma unroll
        for (int p = 0; p < 8; ++p) {
            float v = c[p] + bias;
            v = v > 0.f ? v : 0.f;
            __builtin_nontemporal_store(f2b(v), &Ab[(size_t)(e0 + p + mo) * HC + c0 + n]);
        }
    }
}

// ================================================================
// CSR incidence build (one-time): node -> list of incident edge ids.
// Reference scatter: msg[e] adds to tgt always, and to src iff s!=t.
// ================================================================
__global__ void kcsr_zero(int* __restrict__ cnt) {
    int i = blockIdx.x * 256 + threadIdx.x;
    if (i < NN) cnt[i] = 0;
}

__global__ void kcsr_count(const int* __restrict__ edges, int* __restrict__ cnt) {
    int e = blockIdx.x * 256 + threadIdx.x;
    if (e >= EE) return;
    int s = edges[2 * e + 0], t = edges[2 * e + 1];
    atomicAdd(&cnt[t], 1);
    if (s != t) atomicAdd(&cnt[s], 1);
}

// One-block exclusive scan over NN counts (256 threads x 8 elems = 2048 >= NN).
__global__ void kcsr_scan(const int* __restrict__ cnt,
                          int* __restrict__ rowptr, int* __restrict__ cursor) {
    __shared__ int part[256];
    const int t = threadIdx.x;
    const int base = t * 8;
    int loc[8];
    int sum = 0;
#pragma unroll
    for (int i = 0; i < 8; ++i) {
        int v = (base + i < NN) ? cnt[base + i] : 0;
        loc[i] = sum;
        sum += v;
    }
    part[t] = sum;
    __syncthreads();
    // Hillis-Steele inclusive scan over partials
    for (int d = 1; d < 256; d <<= 1) {
        int v = (t >= d) ? part[t - d] : 0;
        __syncthreads();
        part[t] += v;
        __syncthreads();
    }
    const int excl = (t == 0) ? 0 : part[t - 1];
#pragma unroll
    for (int i = 0; i < 8; ++i) {
        if (base + i < NN) {
            rowptr[base + i] = excl + loc[i];
            cursor[base + i] = excl + loc[i];
        }
    }
    if (t == 255) rowptr[NN] = part[255];
}

__global__ void kcsr_fill(const int* __restrict__ edges,
                          int* __restrict__ cursor, int* __restrict__ lists) {
    int e = blockIdx.x * 256 + threadIdx.x;
    if (e >= EE) return;
    int s = edges[2 * e + 0], t = edges[2 * e + 1];
    int p = atomicAdd(&cursor[t], 1);
    lists[p] = e;
    if (s != t) {
        p = atomicAdd(&cursor[s], 1);
        lists[p] = e;
    }
}

// ================================================================
// K3 (per step): msg[e] = A[e] @ x[src[e]], plain coalesced stores.
// One wave per edge; lane owns rows 2*lane, 2*lane+1 (one 128B A line
// each). A loads are non-temporal (pure stream, 524MB > L2).
// ================================================================
__global__ void k3_message(const unsigned short* __restrict__ Ab,
                           const float* __restrict__ x,
                           const int*   __restrict__ edges,
                           float* __restrict__ msg) {
    __shared__ float xs[4][HH];
    const int wave = threadIdx.x >> 5;
    const int lane = threadIdx.x & 31;
    const int e = blockIdx.x * 4 + wave;
    if (e >= EE) return;

    const int s = edges[2 * e + 0];
    const float* xp = x + (size_t)s * HH;
    xs[wave][lane]      = xp[lane];
    xs[wave][lane + 32] = xp[lane + 32];
    __syncthreads();

    const u32x4* Ap = (const u32x4*)(Ab + (size_t)e * HC);
    float acc[2];
#pragma unroll
    for (int half = 0; half < 2; ++half) {
        const int r = lane * 2 + half;
        const u32x4* rp = Ap + r * 8;       // 64 bf16 = 8 x 16B = one 128B line
        float a = 0.f;
#pragma unroll
        for (int q = 0; q < 8; ++q) {
            u32x4 d = __builtin_nontemporal_load(rp + q);
            const float* xv = &xs[wave][q * 8];
            a += b2f_lo(d[0]) * xv[0] + b2f_hi(d[0]) * xv[1]
               + b2f_lo(d[1]) * xv[2] + b2f_hi(d[1]) * xv[3]
               + b2f_lo(d[2]) * xv[4] + b2f_hi(d[2]) * xv[5]
               + b2f_lo(d[3]) * xv[6] + b2f_hi(d[3]) * xv[7];
        }
        acc[half] = a;
    }
    // rows 2*lane, 2*lane+1 -> one 8B store; wave covers 256B contiguous
    float2 out; out.x = acc[0]; out.y = acc[1];
    *(float2*)(msg + (size_t)e * HH + lane * 2) = out;
}

// ================================================================
// K4 (per step): fused gather (CSR sum of msg rows) + GRU cell.
// One block (64 threads) per node.
// ================================================================
__global__ void k4_gru(const float* __restrict__ x,
                       const float* __restrict__ msg,
                       const int*   __restrict__ rowptr,
                       const int*   __restrict__ lists,
                       const float* __restrict__ Wih,  // 128 x 192
                       const float* __restrict__ Whh,  //  64 x 192
                       const float* __restrict__ bih,
                       const float* __restrict__ bhh,
                       float* __restrict__ xout) {
    __shared__ float inp[2 * HH];           // [x ; m]
    const int node = blockIdx.x;
    const int t = threadIdx.x;              // 0..63
    const float xv = x[(size_t)node * HH + t];

    // gather m[node][t] = sum over incident edges of msg[e][t]
    const int beg = rowptr[node];
    const int end = rowptr[node + 1];
    float mv = 0.f;
    int idx = beg;
    for (; idx + 1 < end; idx += 2) {
        int e0 = lists[idx], e1 = lists[idx + 1];
        mv += msg[(size_t)e0 * HH + t] + msg[(size_t)e1 * HH + t];
    }
    if (idx < end) mv += msg[(size_t)lists[idx] * HH + t];

    inp[t]      = xv;
    inp[HH + t] = mv;
    __syncthreads();

    float gi[3], gh[3];
#pragma unroll
    for (int g = 0; g < 3; ++g) {
        const int c = g * HH + t;
        float a = bih[c];
#pragma unroll 8
        for (int k = 0; k < 2 * HH; ++k) a += inp[k] * Wih[k * 192 + c];
        gi[g] = a;
        float h = bhh[c];
#pragma unroll 8
        for (int k = 0; k < HH; ++k) h += inp[k] * Whh[k * 192 + c];
        gh[g] = h;
    }
    const float r = 1.f / (1.f + expf(-(gi[0] + gh[0])));
    const float z = 1.f / (1.f + expf(-(gi[1] + gh[1])));
    const float n = tanhf(gi[2] + r * gh[2]);
    xout[(size_t)node * HH + t] = (1.f - z) * n + z * xv;
}

// ================================================================
// Host launcher
// ================================================================
extern "C" void kernel_launch(void* const* d_in, const int* in_sizes, int n_in,
                              void* d_out, int out_size, void* d_ws, size_t ws_size,
                              hipStream_t stream) {
    const float* x_in      = (const float*)d_in[0];
    // d_in[1] = adj (unused by reference math)
    const float* edge_data = (const float*)d_in[2];
    const int*   edges     = (const int*)  d_in[3];
    // d_in[4] = T (value 8, hardcoded per reference constant)
    const float* W1  = (const float*)d_in[5];
    const float* b1  = (const float*)d_in[6];
    const float* W2  = (const float*)d_in[7];
    const float* b2  = (const float*)d_in[8];
    const float* Wih = (const float*)d_in[9];
    const float* Whh = (const float*)d_in[10];
    const float* bih = (const float*)d_in[11];
    const float* bhh = (const float*)d_in[12];

    // ---- workspace layout (all offsets 256B-aligned) ----
    char* ws = (char*)d_ws;
    size_t off = 0;
    unsigned short* Ab   = (unsigned short*)(ws + off); off += (size_t)EE * HC * 2;     // 524 MB
    unsigned short* h1b  = (unsigned short*)(ws + off); off += (size_t)EE * ENHH * 2;   // 16.4 MB
    unsigned short* w2s  = (unsigned short*)(ws + off); off += (size_t)ENHH * HC * 2;   // 1 MB
    float* msg  = (float*)(ws + off); off += (size_t)EE * HH * 4;                       // 16.4 MB
    float* xb0  = (float*)(ws + off); off += (size_t)NN * HH * 4;
    float* xb1  = (float*)(ws + off); off += (size_t)NN * HH * 4;
    int* cnt    = (int*)(ws + off); off += 8192;
    int* rowptr = (int*)(ws + off); off += 8192;
    int* cursor = (int*)(ws + off); off += 8192;
    int* lists  = (int*)(ws + off); off += (size_t)2 * EE * 4;                          // 512 KB
    (void)ws_size; (void)in_sizes; (void)n_in; (void)out_size;

    hipMemcpyAsync(xb0, x_in, (size_t)NN * HH * 4, hipMemcpyDeviceToDevice, stream);

    // ---- one-time: CSR incidence build ----
    kcsr_zero <<<(NN + 255) / 256, 256, 0, stream>>>(cnt);
    kcsr_count<<<(EE + 255) / 256, 256, 0, stream>>>(edges, cnt);
    kcsr_scan <<<1, 256, 0, stream>>>(cnt, rowptr, cursor);
    kcsr_fill <<<(EE + 255) / 256, 256, 0, stream>>>(edges, cursor, lists);

    // ---- one-time: edge network ----
    kw2_swizzle<<<(256 * 4 * 32 * 16) / 256, 256, 0, stream>>>(W2, w2s);
    k1_h1<<<EE, ENHH, 0, stream>>>(edge_data, edges, W1, b1, h1b);
    k2_edge_matrices<<<EE / 16, 128, 0, stream>>>(h1b, w2s, b2, Ab);

    // ---- T message-passing steps ----
    for (int step = 0; step < TSTEPS; ++step) {
        float* xc = (step & 1) ? xb1 : xb0;
        float* xn = (step & 1) ? xb0 : xb1;
        k3_message<<<EE / 4, 128, 0, stream>>>(Ab, xc, edges, msg);
        k4_gru<<<NN, HH, 0, stream>>>(xc, msg, rowptr, lists, Wih, Whh, bih, bhh, xn);
    }

    // T=8 (even) -> result in xb0
    hipMemcpyAsync(d_out, xb0, (size_t)NN * HH * 4, hipMemcpyDeviceToDevice, stream);
}
